// MixtralSparseMoeBlock_43516608643301
// MI455X (gfx1250) — matrix operand order repaired and checked by reading, not compiled, not run on hardware
//
#include <hip/hip_runtime.h>
#include <hip/hip_bf16.h>

// ---- problem constants (B=2,S=2048 -> T=4096; H=1024, F=2048, E=8, top-k=2) ----
#define T_TOK 4096
#define H_DIM 1024
#define F_DIM 2048
#define E_NUM 8

typedef __attribute__((ext_vector_type(16))) __bf16 v16bf;
typedef __attribute__((ext_vector_type(8)))  float  v8f;

union Frag {                 // one WMMA 16x32 bf16 operand: 8 VGPRs / lane
  v16bf v;
  uint4 q[2];
};

// round-to-nearest-even fp32 -> bf16 (epilogue/router only)
__device__ __forceinline__ unsigned short f2bf(float f) {
  unsigned int u = __float_as_uint(f);
  u += 0x7FFFu + ((u >> 16) & 1u);
  return (unsigned short)(u >> 16);
}
// single v_perm_b32: pack {bf16(hi), bf16(lo)} by truncation (weights staging)
__device__ __forceinline__ unsigned pack_bf16(float lo, float hi) {
  return __builtin_amdgcn_perm(__float_as_uint(hi), __float_as_uint(lo), 0x07060302u);
}

// async copy of 32 bytes / lane: global (saddr + voffset) -> LDS, tracked by ASYNCcnt.
// inst_offset applies to BOTH the LDS and global addresses, so the second 16-B
// chunk reuses the same address registers with offset:16.
__device__ __forceinline__ void async_copy32(unsigned lds_addr, unsigned gvoff,
                                             const void* sbase) {
  asm volatile("global_load_async_to_lds_b128 %0, %1, %2 offset:0"
               :: "v"(lds_addr), "v"(gvoff),
                  "s"((unsigned long long)(size_t)sbase) : "memory");
  asm volatile("global_load_async_to_lds_b128 %0, %1, %2 offset:16"
               :: "v"(lds_addr), "v"(gvoff),
                  "s"((unsigned long long)(size_t)sbase) : "memory");
}
__device__ __forceinline__ void wait_async0() {
  asm volatile("s_wait_asynccnt 0x0" ::: "memory");
}
__device__ __forceinline__ unsigned lds_off(const void* p) {
  return (unsigned)(size_t)p;            // flat LDS addr = {aperture, offset[31:0]}
}

// ---------------------------------------------------------------------------
// Router: one wave per token. After top-2 renormalization only the two largest
// logits matter: w0 = 1/(1+exp(l1-l0)), w1 = 1-w0.
// ---------------------------------------------------------------------------
__global__ void moe_router(const float* __restrict__ x,
                           const float* __restrict__ gw,
                           float* __restrict__ combine) {
  const int wid  = threadIdx.x >> 5;
  const int lane = threadIdx.x & 31;
  const int t    = blockIdx.x * 8 + wid;

  float acc[E_NUM];
#pragma unroll
  for (int e = 0; e < E_NUM; ++e) acc[e] = 0.f;

  const float* xr = x + (size_t)t * H_DIM;
  for (int h = lane; h < H_DIM; h += 32) {
    const float xv = xr[h];
    const float* g = gw + (size_t)h * E_NUM;
#pragma unroll
    for (int e = 0; e < E_NUM; ++e) acc[e] += xv * g[e];
  }
#pragma unroll
  for (int off = 16; off >= 1; off >>= 1) {
#pragma unroll
    for (int e = 0; e < E_NUM; ++e) acc[e] += __shfl_xor(acc[e], off, 32);
  }
  if (lane == 0) {
    int i0 = 0;
#pragma unroll
    for (int e = 1; e < E_NUM; ++e) if (acc[e] > acc[i0]) i0 = e;
    int i1 = (i0 == 0) ? 1 : 0;
#pragma unroll
    for (int e = 0; e < E_NUM; ++e) if (e != i0 && acc[e] > acc[i1]) i1 = e;
    const float w0 = __builtin_amdgcn_rcpf(1.f + __expf(acc[i1] - acc[i0]));
    const float w1 = 1.f - w0;
    float* c = combine + (size_t)t * E_NUM;
#pragma unroll
    for (int e = 0; e < E_NUM; ++e) c[e] = (e == i0) ? w0 : ((e == i1) ? w1 : 0.f);
  }
}

// fp32 -> bf16 bulk convert (4 elements / thread, v_perm packing)
__global__ void moe_cvt_bf16(const float* __restrict__ in,
                             unsigned* __restrict__ out, int n4) {
  const int i = blockIdx.x * blockDim.x + threadIdx.x;
  if (i >= n4) return;
  const float4 f = ((const float4*)in)[i];
  uint2 o;
  o.x = pack_bf16(f.x, f.y);
  o.y = pack_bf16(f.z, f.w);
  ((uint2*)out)[i] = o;
}

#define KT 32
#define LDSW 40   // padded K stride (ushorts); 80 B rows keep 16-B alignment

// ---------------------------------------------------------------------------
// GEMM1 (per expert): P[t,f] = bf16( silu(X@Wu)[t,f] * (X@Wg)[t,f] )
//  X: [T,H] bf16, Wu/Wg: [H,F] fp32 (packed to bf16 per tile)
//  block tile 128(M) x 64(N), K-tile 32; waves 4x2; wave tile 32x32
//  Double-buffered LDS: async A-DMA + weight loads for tile i+1 overlap WMMA(i).
// ---------------------------------------------------------------------------
__global__ void __launch_bounds__(256)
moe_gemm1(const unsigned short* __restrict__ X,
          const float* __restrict__ Wu,
          const float* __restrict__ Wg,
          unsigned short* __restrict__ P) {
  __shared__ unsigned short lA [2][128][LDSW];
  __shared__ unsigned short lBu[2][ 64][LDSW];
  __shared__ unsigned short lBg[2][ 64][LDSW];

  const int n0 = blockIdx.x * 64;     // F tile base
  const int m0 = blockIdx.y * 128;    // token tile base
  const int wid  = threadIdx.x >> 5;
  const int lane = threadIdx.x & 31;
  const int waveM = wid & 3;
  const int waveN = wid >> 2;
  const int lm = lane & 15;
  const int hi = lane >> 4;
  const int ka = hi * 8;              // A: K halves 0-7/16-23 vs 8-15/24-31
  const int kb = hi * 8;              // B: K halves 0-7/16-23 vs 8-15/24-31

  const int arow = threadIdx.x >> 1;
  const int acol = (threadIdx.x & 1) * 16;
  const unsigned aLds[2] = { lds_off(&lA[0][arow][acol]),
                             lds_off(&lA[1][arow][acol]) };
  const int kp = (threadIdx.x >> 4) << 1;   // even K row: 0,2,..,30
  const int nq = (threadIdx.x & 15) << 2;   // 0..60 step 4

  auto stage = [&](int b, int k0) {
    async_copy32(aLds[b], (unsigned)((size_t)(m0 + arow) * H_DIM + k0 + acol) * 2u, X);
    const float4 u0 = *(const float4*)&Wu[(size_t)(k0 + kp    ) * F_DIM + n0 + nq];
    const float4 u1 = *(const float4*)&Wu[(size_t)(k0 + kp + 1) * F_DIM + n0 + nq];
    const float4 g0 = *(const float4*)&Wg[(size_t)(k0 + kp    ) * F_DIM + n0 + nq];
    const float4 g1 = *(const float4*)&Wg[(size_t)(k0 + kp + 1) * F_DIM + n0 + nq];
    *(unsigned*)&lBu[b][nq + 0][kp] = pack_bf16(u0.x, u1.x);
    *(unsigned*)&lBu[b][nq + 1][kp] = pack_bf16(u0.y, u1.y);
    *(unsigned*)&lBu[b][nq + 2][kp] = pack_bf16(u0.z, u1.z);
    *(unsigned*)&lBu[b][nq + 3][kp] = pack_bf16(u0.w, u1.w);
    *(unsigned*)&lBg[b][nq + 0][kp] = pack_bf16(g0.x, g1.x);
    *(unsigned*)&lBg[b][nq + 1][kp] = pack_bf16(g0.y, g1.y);
    *(unsigned*)&lBg[b][nq + 2][kp] = pack_bf16(g0.z, g1.z);
    *(unsigned*)&lBg[b][nq + 3][kp] = pack_bf16(g0.w, g1.w);
    if (k0 + KT < H_DIM) {                 // prefetch next K-tile of the weight stream
      __builtin_prefetch(&Wu[(size_t)(k0 + KT + kp) * F_DIM + n0 + nq], 0, 1);
      __builtin_prefetch(&Wg[(size_t)(k0 + KT + kp) * F_DIM + n0 + nq], 0, 1);
    }
  };

  v8f accU[2][2] = {};
  v8f accG[2][2] = {};

  stage(0, 0);
  const int NIT = H_DIM / KT;
  for (int it = 0; it < NIT; ++it) {
    const int cur = it & 1;
    wait_async0();                         // this wave's async writes to LDS done
    __syncthreads();                       // everyone's staging for tile `it` visible
    if (it + 1 < NIT) stage(cur ^ 1, (it + 1) * KT);

    Frag a[2], bu[2], bg[2];
#pragma unroll
    for (int i = 0; i < 2; ++i) {
      const int m = waveM * 32 + i * 16 + lm;
      a[i].q[0] = *(const uint4*)&lA[cur][m][ka];
      a[i].q[1] = *(const uint4*)&lA[cur][m][ka + 16];
    }
#pragma unroll
    for (int j = 0; j < 2; ++j) {
      const int n = waveN * 32 + j * 16 + lm;
      bu[j].q[0] = *(const uint4*)&lBu[cur][n][kb];
      bu[j].q[1] = *(const uint4*)&lBu[cur][n][kb + 16];
      bg[j].q[0] = *(const uint4*)&lBg[cur][n][kb];
      bg[j].q[1] = *(const uint4*)&lBg[cur][n][kb + 16];
    }
#pragma unroll
    for (int i = 0; i < 2; ++i)
#pragma unroll
      for (int j = 0; j < 2; ++j) {
        accU[i][j] = __builtin_amdgcn_wmma_f32_16x16x32_bf16(
            false, a[i].v, false, bu[j].v, (short)0, accU[i][j], false, false);
        accG[i][j] = __builtin_amdgcn_wmma_f32_16x16x32_bf16(
            false, a[i].v, false, bg[j].v, (short)0, accG[i][j], false, false);
      }
  }

  // ---- epilogue: silu(up) * gate -> bf16 P (v_rcp, v_exp) ----
#pragma unroll
  for (int i = 0; i < 2; ++i)
#pragma unroll
    for (int j = 0; j < 2; ++j)
#pragma unroll
      for (int v = 0; v < 8; ++v) {
        const int tr = m0 + waveM * 32 + i * 16 + hi * 8 + v;
        const int fc = n0 + waveN * 32 + j * 16 + lm;
        const float up = accU[i][j][v];
        const float gt = accG[i][j][v];
        const float s  = up * __builtin_amdgcn_rcpf(1.f + __expf(-up));
        P[(size_t)tr * F_DIM + fc] = f2bf(s * gt);
      }
}

// ---------------------------------------------------------------------------
// GEMM2 (per expert): out[t,h] (+)= combine[t,e] * (P @ Wd)[t,h]
//  P: [T,F] bf16, Wd: [F,H] fp32.
//  block tile 128(M) x 128(N), K-tile 32; waves 4x2; wave tile 32x64
// ---------------------------------------------------------------------------
__global__ void __launch_bounds__(256)
moe_gemm2(const unsigned short* __restrict__ P,
          const float* __restrict__ Wd,
          const float* __restrict__ combine,
          float* __restrict__ out,
          int expert, int addTo) {
  __shared__ unsigned short lA[2][128][LDSW];
  __shared__ unsigned short lB[2][128][LDSW];

  const int n0 = blockIdx.x * 128;    // H tile base
  const int m0 = blockIdx.y * 128;    // token tile base
  const int wid  = threadIdx.x >> 5;
  const int lane = threadIdx.x & 31;
  const int waveM = wid & 3;          // 32 rows each
  const int waveN = wid >> 2;         // 64 cols each
  const int lm = lane & 15;
  const int hi = lane >> 4;
  const int ka = hi * 8;
  const int kb = hi * 8;

  const int arow = threadIdx.x >> 1;
  const int acol = (threadIdx.x & 1) * 16;
  const unsigned aLds[2] = { lds_off(&lA[0][arow][acol]),
                             lds_off(&lA[1][arow][acol]) };

  auto stage = [&](int b, int k0) {
    async_copy32(aLds[b], (unsigned)((size_t)(m0 + arow) * F_DIM + k0 + acol) * 2u, P);
    // 32(K) x 128(N) fp32 -> packed bf16 [N][K]; 512 K-pair quads, 2 per thread
#pragma unroll
    for (int r = 0; r < 2; ++r) {
      const int idx = threadIdx.x + r * 256;
      const int kp = (idx >> 5) << 1;          // even K row 0..30
      const int nq = (idx & 31) << 2;          // 0..124 step 4
      const float4 d0 = *(const float4*)&Wd[(size_t)(k0 + kp    ) * H_DIM + n0 + nq];
      const float4 d1 = *(const float4*)&Wd[(size_t)(k0 + kp + 1) * H_DIM + n0 + nq];
      *(unsigned*)&lB[b][nq + 0][kp] = pack_bf16(d0.x, d1.x);
      *(unsigned*)&lB[b][nq + 1][kp] = pack_bf16(d0.y, d1.y);
      *(unsigned*)&lB[b][nq + 2][kp] = pack_bf16(d0.z, d1.z);
      *(unsigned*)&lB[b][nq + 3][kp] = pack_bf16(d0.w, d1.w);
      if (k0 + KT < F_DIM)
        __builtin_prefetch(&Wd[(size_t)(k0 + KT + kp) * H_DIM + n0 + nq], 0, 1);
    }
  };

  v8f acc[2][4] = {};

  stage(0, 0);
  const int NIT = F_DIM / KT;
  for (int it = 0; it < NIT; ++it) {
    const int cur = it & 1;
    wait_async0();
    __syncthreads();
    if (it + 1 < NIT) stage(cur ^ 1, (it + 1) * KT);

    Frag a[2], b[4];
#pragma unroll
    for (int i = 0; i < 2; ++i) {
      const int m = waveM * 32 + i * 16 + lm;
      a[i].q[0] = *(const uint4*)&lA[cur][m][ka];
      a[i].q[1] = *(const uint4*)&lA[cur][m][ka + 16];
    }
#pragma unroll
    for (int j = 0; j < 4; ++j) {
      const int n = waveN * 64 + j * 16 + lm;
      b[j].q[0] = *(const uint4*)&lB[cur][n][kb];
      b[j].q[1] = *(const uint4*)&lB[cur][n][kb + 16];
    }
#pragma unroll
    for (int i = 0; i < 2; ++i)
#pragma unroll
      for (int j = 0; j < 4; ++j)
        acc[i][j] = __builtin_amdgcn_wmma_f32_16x16x32_bf16(
            false, a[i].v, false, b[j].v, (short)0, acc[i][j], false, false);
  }

  // ---- epilogue: scale rows by combine[t, expert], write/accumulate fp32 ----
#pragma unroll
  for (int i = 0; i < 2; ++i)
#pragma unroll
    for (int v = 0; v < 8; ++v) {
      const int tr = m0 + waveM * 32 + i * 16 + hi * 8 + v;
      const float c = combine[(size_t)tr * E_NUM + expert];
#pragma unroll
      for (int j = 0; j < 4; ++j) {
        const int hc = n0 + waveN * 64 + j * 16 + lm;
        const size_t o = (size_t)tr * H_DIM + hc;
        const float val = c * acc[i][j][v];
        out[o] = addTo ? (out[o] + val) : val;
      }
    }
}

// ---------------------------------------------------------------------------
extern "C" void kernel_launch(void* const* d_in, const int* in_sizes, int n_in,
                              void* d_out, int out_size, void* d_ws, size_t ws_size,
                              hipStream_t stream) {
  (void)in_sizes; (void)n_in; (void)out_size; (void)ws_size;
  const float* x      = (const float*)d_in[0];   // [T, H]
  const float* gate_w = (const float*)d_in[1];   // [H, E]
  const float* w_up   = (const float*)d_in[2];   // [E, H, F]
  const float* w_gate = (const float*)d_in[3];   // [E, H, F]
  const float* w_down = (const float*)d_in[4];   // [E, F, H]
  float* out = (float*)d_out;                    // [T, H]

  // workspace layout
  char* ws = (char*)d_ws;
  float* combine        = (float*)ws;                                   // T*E fp32   (128 KB)
  unsigned short* xb    = (unsigned short*)(ws + 256 * 1024);           // T*H bf16   (8 MB)
  unsigned short* Pbuf  = (unsigned short*)(ws + 256 * 1024 +
                                            (size_t)T_TOK * H_DIM * 2); // T*F bf16   (16 MB)

  moe_router<<<T_TOK / 8, 256, 0, stream>>>(x, gate_w, combine);
  moe_cvt_bf16<<<(T_TOK * H_DIM / 4) / 256, 256, 0, stream>>>(x, (unsigned*)xb,
                                                              T_TOK * H_DIM / 4);

  const dim3 g1(F_DIM / 64, T_TOK / 128);
  const dim3 g2(H_DIM / 128, T_TOK / 128);
  for (int e = 0; e < E_NUM; ++e) {
    moe_gemm1<<<g1, 256, 0, stream>>>(xb,
                                      w_up   + (size_t)e * H_DIM * F_DIM,
                                      w_gate + (size_t)e * H_DIM * F_DIM,
                                      Pbuf);
    moe_gemm2<<<g2, 256, 0, stream>>>(Pbuf,
                                      w_down + (size_t)e * F_DIM * H_DIM,
                                      combine, out, e, e != 0);
  }
}
